// FrameLevelPartFeatureExtractor_44409961840672
// MI455X (gfx1250) — matrix operand order repaired and checked by reading, not compiled, not run on hardware
//
#include <hip/hip_runtime.h>
#include <hip/hip_bf16.h>

// ---------------------------------------------------------------------------
// CDNA5 (gfx1250, wave32) implicit-GEMM conv pipeline using v_wmma_f32_16x16x32_f16.
// Activations: f16 NHWC with 1-column zero halo (pitch W+2) -> no boundary
// predication in the WMMA inner loop. Weights: f16 [tap][oc][cin].
// Each wave: 16 oc x 64 w (4 N-subtiles); A fragment reused by 4 wmmas;
// B fragments batch-loaded ahead of the wmma burst to overlap VMEM latency.
// Double leaky_relu(0.01) folds to: y<0 ? 1e-4*y : y.
// ---------------------------------------------------------------------------

typedef __attribute__((ext_vector_type(16))) _Float16 v16h;
typedef __attribute__((ext_vector_type(8)))  _Float16 half8;
typedef __attribute__((ext_vector_type(8)))  float    v8f;

union AFrag { v16h v; half8 h[2]; };

__device__ __forceinline__ float dlrelu(float v) {
    return v < 0.0f ? v * 1.0e-4f : v;   // leaky_relu(0.01) applied twice
}

// ---------------------------------------------------------------------------
// Weight prep: OIHW fp32 -> [tap][o][c] f16   (tap = kh*3+kw)
// ---------------------------------------------------------------------------
__global__ __launch_bounds__(256)
void prep_weights_f16(const float* __restrict__ in, _Float16* __restrict__ out,
                      int O, int C) {
    int idx = blockIdx.x * 256 + threadIdx.x;
    int total = O * C * 9;
    if (idx >= total) return;
    int o   = idx / (C * 9);
    int rem = idx % (C * 9);
    int c   = rem / 9;
    int tap = rem % 9;
    out[(tap * O + o) * C + c] = (_Float16)in[idx];   // in[((o*C)+c)*9 + tap] == in[idx]
}

// ---------------------------------------------------------------------------
// Layer 1: direct conv, C_IN=1, 5x5, pad 2, slice 512 (0.8 GFLOP = 3% of work).
// NCHW f32 -> padded NHWC f16 [2][1024][258][32], halo cols zeroed here.
// ---------------------------------------------------------------------------
__global__ __launch_bounds__(256)
void conv1_5x5(const float* __restrict__ x, const float* __restrict__ w1,
               _Float16* __restrict__ out) {
    constexpr int H = 1024, W = 256, WS = W + 2, SH = 512, OC = 32;
    __shared__ float ws[OC * 25];
    for (int i = threadIdx.x; i < OC * 25; i += 256) ws[i] = w1[i];
    __syncthreads();

    long idx = (long)blockIdx.x * 256 + threadIdx.x;
    if (idx >= 2L * H * W) return;
    int w = (int)(idx % W);
    int h = (int)((idx / W) % H);
    int b = (int)(idx / ((long)W * H));
    int hs = h % SH;

    float s[OC];
#pragma unroll
    for (int o = 0; o < OC; ++o) s[o] = 0.0f;

    for (int kh = 0; kh < 5; ++kh) {
        int ihs = hs + kh - 2;
        if (ihs < 0 || ihs >= SH) continue;
        int ih = h + kh - 2;
        for (int kw = 0; kw < 5; ++kw) {
            int iw = w + kw - 2;
            if (iw < 0 || iw >= W) continue;
            float xv = x[((long)b * H + ih) * W + iw];
#pragma unroll
            for (int o = 0; o < OC; ++o)
                s[o] = fmaf(xv, ws[o * 25 + kh * 5 + kw], s[o]);
        }
    }
    long rowbase = (long)(b * H + h) * WS;
    _Float16* op = out + (rowbase + w + 1) * OC;
    half8 z = {};
#pragma unroll
    for (int g = 0; g < OC / 8; ++g) {
        half8 hv;
#pragma unroll
        for (int r = 0; r < 8; ++r) hv[r] = (_Float16)dlrelu(s[g * 8 + r]);
        *(half8*)(op + g * 8) = hv;
    }
    if (w == 0) {          // zero left halo column
        _Float16* hp = out + rowbase * OC;
#pragma unroll
        for (int g = 0; g < OC / 8; ++g) *(half8*)(hp + g * 8) = z;
    }
    if (w == W - 1) {      // zero right halo column
        _Float16* hp = out + (rowbase + W + 1) * OC;
#pragma unroll
        for (int g = 0; g < OC / 8; ++g) *(half8*)(hp + g * 8) = z;
    }
}

// ---------------------------------------------------------------------------
// 2x2 maxpool on halo-padded NHWC f16 (pitch Win+2 -> pitch OW+2), zeroing
// the output halo columns.
// ---------------------------------------------------------------------------
template <int C>
__global__ __launch_bounds__(256)
void maxpool2_nhwc(const _Float16* __restrict__ in, _Float16* __restrict__ out,
                   int B, int Hin, int Win) {
    int OH = Hin >> 1, OW = Win >> 1;
    int WSi = Win + 2, WSo = OW + 2;
    long idx = (long)blockIdx.x * 256 + threadIdx.x;
    long total = (long)B * OH * OW * C;
    if (idx >= total) return;
    int c  = (int)(idx % C);
    int ow = (int)((idx / C) % OW);
    int oh = (int)((idx / ((long)C * OW)) % OH);
    int b  = (int)(idx / ((long)C * OW * OH));
    const _Float16* p = in + (((long)b * Hin + 2 * oh) * WSi + 2 * ow + 1) * C + c;
    float m0 = (float)p[0];
    float m1 = (float)p[C];
    float m2 = (float)p[(long)WSi * C];
    float m3 = (float)p[(long)WSi * C + C];
    float m = fmaxf(fmaxf(m0, m1), fmaxf(m2, m3));
    long orow = (long)(b * OH + oh) * WSo;
    out[(orow + ow + 1) * C + c] = (_Float16)m;
    if (ow == 0)      out[orow * C + c]            = (_Float16)0.0f;
    if (ow == OW - 1) out[(orow + OW + 1) * C + c] = (_Float16)0.0f;
}

// ---------------------------------------------------------------------------
// WMMA implicit-GEMM 3x3 conv, pad 1, focal slice height SH (compile-time).
//   Wave tile: 16 oc x 64 w (4 N-subtiles) at one (b, oh).
//   A (16x32 f16) = weights [tap][oc][cin] : two 16B loads / K-chunk,
//                   reused by 4 wmmas.
//   B (32x16 f16) = halo-padded NHWC acts  : 4 unconditional 32B loads off one
//                   base address with immediate offsets st*16*CIN, issued as a
//                   batch BEFORE the 4-wmma burst (latency overlap).
// FINAL=true: write fp32 NCHW to d_out; else padded f16 NHWC (+ halo zeroing).
// ---------------------------------------------------------------------------
template <int CIN, int COUT, int H, int W, int SH, bool FINAL>
__global__ __launch_bounds__(256)
void conv3x3_wmma(const _Float16* __restrict__ xin, const _Float16* __restrict__ wf,
                  _Float16* __restrict__ outh, float* __restrict__ outf) {
    constexpr int B    = 2;
    constexpr int WS   = W + 2;
    constexpr int nOCT = COUT / 16;
    constexpr int nWT  = W / 64;
    constexpr int NST  = 4;                 // N-subtiles per wave

    int tile = blockIdx.x * (256 / 32) + (threadIdx.x >> 5);
    if (tile >= B * H * nWT * nOCT) return;

    int oct = tile % nOCT;
    int t1  = tile / nOCT;
    int wt  = t1 % nWT;
    int t2  = t1 / nWT;
    int oh  = t2 % H;
    int b   = t2 / H;

    int lane  = threadIdx.x & 31;
    int ln15  = lane & 15;
    int khalf = lane >> 4;
    int wbase = wt * 64;
    int ohs   = oh % SH;
    int o     = oct * 16 + ln15;

    v8f acc[NST] = {};

    for (int kh = 0; kh < 3; ++kh) {
        int ihs = ohs + kh - 1;
        if (ihs < 0 || ihs >= SH) continue;  // focal slice boundary (SH=1 -> only kh==1)
        int ih = oh + kh - 1;
        const _Float16* xrow = xin + ((long)(b * H + ih) * WS) * CIN;
        __builtin_prefetch(xrow + (long)WS * CIN, 0, 1);   // gfx1250 global_prefetch_b8
#pragma unroll
        for (int kw = 0; kw < 3; ++kw) {
            // Padded index: logical iw = col + kw - 1, stored at col + kw. Always in range.
            const _Float16* xcol = xrow + (long)(wbase + ln15 + kw) * CIN + khalf * 16;
            const _Float16* wp0  = wf + ((long)((kh * 3 + kw) * COUT + o)) * CIN + khalf * 16;
#pragma unroll
            for (int c0 = 0; c0 < CIN; c0 += 32) {
                // A fragment (contiguous K, matching B): lane<16 K 0..15;
                // lane>=16 K 16..31.
                AFrag af;
                af.h[0] = *(const half8*)(wp0 + c0);
                af.h[1] = *(const half8*)(wp0 + c0 + 8);
                // Load phase: 4 B fragments into distinct registers (clause of 8
                // b128 loads), then compute phase: 4 wmmas draining the loads in
                // order -> s_wait_loadcnt 6/4/2/0 instead of 0 before every wmma.
                v16h bv[NST];
#pragma unroll
                for (int st = 0; st < NST; ++st)
                    bv[st] = *(const v16h*)(xcol + c0 + st * 16 * CIN);
#pragma unroll
                for (int st = 0; st < NST; ++st)
                    acc[st] = __builtin_amdgcn_wmma_f32_16x16x32_f16(
                        false, af.v, false, bv[st], (short)0, acc[st], false, false);
            }
        }
    }

    // D layout: lane -> col N = ln15; VGPR r -> row M = r + khalf*8.
    int ocb = oct * 16 + khalf * 8;
    if (!FINAL) {
        long rowbase = (long)(b * H + oh) * WS;
        half8 z = {};
#pragma unroll
        for (int st = 0; st < NST; ++st) {
            half8 hv;
#pragma unroll
            for (int r = 0; r < 8; ++r) hv[r] = (_Float16)dlrelu(acc[st][r]);
            int wcol = wbase + st * 16 + ln15;
            *(half8*)(outh + (rowbase + wcol + 1) * COUT + ocb) = hv;
        }
        // zero halo columns of this (oh, oc-slice)
        if (wbase == 0 && ln15 == 0)
            *(half8*)(outh + rowbase * COUT + ocb) = z;
        if (wbase + 64 == W && ln15 == 15)
            *(half8*)(outh + (rowbase + W + 1) * COUT + ocb) = z;
    } else {
#pragma unroll
        for (int st = 0; st < NST; ++st) {
            int wcol = wbase + st * 16 + ln15;
#pragma unroll
            for (int r = 0; r < 8; ++r) {
                int oc = ocb + r;
                outf[(((long)b * COUT + oc) * H + oh) * W + wcol] = dlrelu(acc[st][r]);
            }
        }
    }
}

// ---------------------------------------------------------------------------
// Host launch
// ---------------------------------------------------------------------------
static inline int cdiv(long a, int b) { return (int)((a + b - 1) / b); }

extern "C" void kernel_launch(void* const* d_in, const int* in_sizes, int n_in,
                              void* d_out, int out_size, void* d_ws, size_t ws_size,
                              hipStream_t stream) {
    const float* x  = (const float*)d_in[0];   // [2,1,1024,256]
    const float* w1 = (const float*)d_in[1];   // [32,1,5,5]
    const float* w2 = (const float*)d_in[2];   // [32,32,3,3]
    const float* w3 = (const float*)d_in[3];   // [64,32,3,3]
    const float* w4 = (const float*)d_in[4];   // [64,64,3,3]
    const float* w5 = (const float*)d_in[5];   // [128,64,3,3]
    const float* w6 = (const float*)d_in[6];   // [128,128,3,3]
    float* out = (float*)d_out;                // [2,128,256,64]

    // Workspace: two 33MB ping-pong halo-padded activation buffers + f16 weights.
    char* ws = (char*)d_ws;
    const size_t BUFBYTES = 34603008;          // >= 2*1024*258*32*2 B
    _Float16* bufA = (_Float16*)ws;
    _Float16* bufB = (_Float16*)(ws + BUFBYTES);
    _Float16* w2f  = (_Float16*)(ws + 2 * BUFBYTES);
    _Float16* w3f  = w2f + 9 * 32 * 32;
    _Float16* w4f  = w3f + 9 * 64 * 32;
    _Float16* w5f  = w4f + 9 * 64 * 64;
    _Float16* w6f  = w5f + 9 * 128 * 64;

    // --- weight conversion (fp32 OIHW -> f16 [tap][o][c]) ---
    prep_weights_f16<<<cdiv(9 * 32 * 32, 256),  256, 0, stream>>>(w2, w2f, 32, 32);
    prep_weights_f16<<<cdiv(9 * 64 * 32, 256),  256, 0, stream>>>(w3, w3f, 64, 32);
    prep_weights_f16<<<cdiv(9 * 64 * 64, 256),  256, 0, stream>>>(w4, w4f, 64, 64);
    prep_weights_f16<<<cdiv(9 * 128 * 64, 256), 256, 0, stream>>>(w5, w5f, 128, 64);
    prep_weights_f16<<<cdiv(9 * 128 * 128, 256),256, 0, stream>>>(w6, w6f, 128, 128);

    // --- L1: 1->32, 5x5, pad2, slice 512. NCHW f32 -> padded NHWC f16 (bufA) ---
    conv1_5x5<<<cdiv(2L * 1024 * 256, 256), 256, 0, stream>>>(x, w1, bufA);

    // --- L2: 32->32, slice 512.  bufA -> bufB.  tiles = 2*1024*4*2 ---
    conv3x3_wmma<32, 32, 1024, 256, 512, false>
        <<<cdiv(2L * 1024 * 4 * 2, 8), 256, 0, stream>>>(bufA, w2f, bufB, nullptr);

    // --- pool1: [2,1024,258,32] -> [2,512,130,32]  bufB -> bufA ---
    maxpool2_nhwc<32><<<cdiv(2L * 512 * 128 * 32, 256), 256, 0, stream>>>(bufB, bufA, 2, 1024, 256);

    // --- L3: 32->64, slice 32.  bufA -> bufB.  tiles = 2*512*2*4 ---
    conv3x3_wmma<32, 64, 512, 128, 32, false>
        <<<cdiv(2L * 512 * 2 * 4, 8), 256, 0, stream>>>(bufA, w3f, bufB, nullptr);

    // --- L4: 64->64, slice 32.  bufB -> bufA ---
    conv3x3_wmma<64, 64, 512, 128, 32, false>
        <<<cdiv(2L * 512 * 2 * 4, 8), 256, 0, stream>>>(bufB, w4f, bufA, nullptr);

    // --- pool2: [2,512,130,64] -> [2,256,66,64]  bufA -> bufB ---
    maxpool2_nhwc<64><<<cdiv(2L * 256 * 64 * 64, 256), 256, 0, stream>>>(bufA, bufB, 2, 512, 128);

    // --- L5: 64->128, slice 1 (effective 1x3).  bufB -> bufA.  tiles = 2*256*1*8 ---
    conv3x3_wmma<64, 128, 256, 64, 1, false>
        <<<cdiv(2L * 256 * 1 * 8, 8), 256, 0, stream>>>(bufB, w5f, bufA, nullptr);

    // --- L6: 128->128, slice 1.  bufA -> d_out (fp32 NCHW) ---
    conv3x3_wmma<128, 128, 256, 64, 1, true>
        <<<cdiv(2L * 256 * 1 * 8, 8), 256, 0, stream>>>(bufA, w6f, nullptr, out);

    (void)in_sizes; (void)n_in; (void)out_size; (void)ws_size;
}
